// MultiHeadAttention_79748952752607
// MI455X (gfx1250) — compile-verified, matrix operand order checked by reading, not run on hardware
//
#include <hip/hip_runtime.h>

// MHA forward for MI455X (gfx1250, wave32, WMMA f16->f32).
// B=2, S=4096, D=768, H=12, DK=64.
//
// Stage 0: convert q,k,v and wq,wk,wv,wo to f16 once (bandwidth-bound, ~5us).
// Stage 1: Q/K/V projections, pure-f16 double-buffered WMMA GEMMs:
//          Qh [B,H,S,DK], Kh [B,H,S,DK], Vt [B,H,DK,S] (transposed for PV).
// Stage 2: causal flash attention, one wave per 16 query rows, online
//          softmax in C-fragment layout, P transposed via per-wave LDS.
// Stage 3: output projection (ctx f16 x wo f16 -> f32 out).
// All intermediates (~93MB) are L2-resident (192MB L2).

#define BB 2
#define SS 4096
#define DD 768
#define HH 12
#define DKK 64
#define BSROWS (BB * SS)   // 8192
#define PPAD 72            // LDS row stride in halves: 144B, 16B-aligned rows

typedef _Float16 half16 __attribute__((ext_vector_type(16)));
typedef float    float8 __attribute__((ext_vector_type(8)));

union HU {                 // one 16x32 f16 WMMA operand (8 VGPRs)
    half16       v;
    unsigned int u[8];
    _Float16     e[16];
};

__device__ __forceinline__ float8 wmma_f16(const HU& a, const HU& b, float8 c) {
    return __builtin_amdgcn_wmma_f32_16x16x32_f16(
        false, a.v, false, b.v, (short)0, c, false, false);
}

// Load one 16x32-f16 A/B fragment for this lane. `base` = row pointer + K-chunk
// offset (halves). ISA 7.12.2 K-pair layout: VGPR0..3 hold K = 8h+{0..7},
// VGPR4..7 hold K = 16+8h+{0..7}  ->  two contiguous 16B loads.
__device__ __forceinline__ void load_frag(HU& f, const _Float16* base, int h) {
    uint4 lo = *(const uint4*)(base + 8 * h);
    uint4 hi = *(const uint4*)(base + 16 + 8 * h);
    f.u[0] = lo.x; f.u[1] = lo.y; f.u[2] = lo.z; f.u[3] = lo.w;
    f.u[4] = hi.x; f.u[5] = hi.y; f.u[6] = hi.z; f.u[7] = hi.w;
}

// ---------------------------------------------------------------------------
// f32 -> f16 conversion pre-pass, 8 elems/thread, b128 in / b128 out.
// ---------------------------------------------------------------------------
__global__ void __launch_bounds__(256)
cvt_kernel(const float* __restrict__ in, _Float16* __restrict__ out, int n)
{
    const int i = (blockIdx.x * 256 + threadIdx.x) * 8;
    if (i >= n) return;
    float4 a = *(const float4*)(in + i);
    float4 b = *(const float4*)(in + i + 4);
    union { _Float16 e[8]; uint4 u; } r;
    r.e[0] = (_Float16)a.x; r.e[1] = (_Float16)a.y;
    r.e[2] = (_Float16)a.z; r.e[3] = (_Float16)a.w;
    r.e[4] = (_Float16)b.x; r.e[5] = (_Float16)b.y;
    r.e[6] = (_Float16)b.z; r.e[7] = (_Float16)b.w;
    *(uint4*)(out + i) = r.u;
}

// ---------------------------------------------------------------------------
// Projection: Y[m,n] = sum_d X[m,d]*W[n,d] + bias[n], all-f16 operands,
// double-buffered K loop. Wave -> 16x64 tile; WG = 8 waves; grid (64, 12).
// vlayout 0: out[((b*H+h)*S+s)*DK+dk]   vlayout 1: out[((b*H+h)*DK+dk)*S+s]
// ---------------------------------------------------------------------------
__global__ void __launch_bounds__(256)
proj_kernel(const _Float16* __restrict__ Xh, const _Float16* __restrict__ Wh,
            const float* __restrict__ bias, _Float16* __restrict__ out,
            int vlayout)
{
    const int lane = threadIdx.x & 31;
    const int wave = threadIdx.x >> 5;
    const int h    = lane >> 4;
    const int nn   = lane & 15;
    const int m0   = blockIdx.x * 128 + wave * 16;
    const int n0   = blockIdx.y * 64;

    const _Float16* xbase = Xh + (size_t)(m0 + nn) * DD;
    const _Float16* wbase[4];
#pragma unroll
    for (int nt = 0; nt < 4; ++nt)
        wbase[nt] = Wh + (size_t)(n0 + nt * 16 + nn) * DD;

    HU a[2], b[2][4];
    float8 acc[4];
#pragma unroll
    for (int nt = 0; nt < 4; ++nt) acc[nt] = (float8)0.0f;

    // prologue: buffer 0 <- kk = 0
    load_frag(a[0], xbase, h);
#pragma unroll
    for (int nt = 0; nt < 4; ++nt) load_frag(b[0][nt], wbase[nt], h);

#pragma unroll 1
    for (int kk = 0; kk < DD; kk += 64) {
        // prefetch buffer 1 <- kk+32 (kk+32 <= 736 < DD always)
        load_frag(a[1], xbase + kk + 32, h);
#pragma unroll
        for (int nt = 0; nt < 4; ++nt)
            load_frag(b[1][nt], wbase[nt] + kk + 32, h);
#pragma unroll
        for (int nt = 0; nt < 4; ++nt)
            acc[nt] = wmma_f16(a[0], b[0][nt], acc[nt]);

        if (kk + 64 < DD) {
            load_frag(a[0], xbase + kk + 64, h);
#pragma unroll
            for (int nt = 0; nt < 4; ++nt)
                load_frag(b[0][nt], wbase[nt] + kk + 64, h);
        }
#pragma unroll
        for (int nt = 0; nt < 4; ++nt)
            acc[nt] = wmma_f16(a[1], b[1][nt], acc[nt]);
    }

#pragma unroll
    for (int nt = 0; nt < 4; ++nt) {
        const int   n  = n0 + nt * 16 + nn;
        const float bv = bias[n];
        const int   hh = n / DKK;
        const int   dk = n % DKK;
#pragma unroll
        for (int r = 0; r < 8; ++r) {
            const int m  = m0 + r + 8 * h;          // C layout: M = r + 8*half
            const int bb = m / SS;
            const int s  = m % SS;
            const float val = acc[nt][r] + bv;
            size_t addr;
            if (vlayout == 0)
                addr = ((size_t)(bb * HH + hh) * SS + s) * DKK + dk;
            else
                addr = ((size_t)(bb * HH + hh) * DKK + dk) * (size_t)SS + s;
            out[addr] = (_Float16)val;
        }
    }
}

// ---------------------------------------------------------------------------
// One 64-key attention block. V fragments loaded before the softmax math so
// global latency overlaps the exp/transpose work. MASKED only for the
// diagonal block.
// ---------------------------------------------------------------------------
template <bool MASKED>
__device__ __forceinline__ void
attn_block(int kb, int q0, int h, int nn,
           const _Float16* __restrict__ Kb, const _Float16* __restrict__ Vb,
           _Float16* ldsP, const HU (&qf)[2],
           float8 (&o)[4], float (&m_run)[8], float (&l_run)[8])
{
    // ---- K fragments (all 8 issued before any WMMA consumes them) ----
    HU kf[4][2];
#pragma unroll
    for (int nt = 0; nt < 4; ++nt)
#pragma unroll
        for (int kc = 0; kc < 2; ++kc)
            load_frag(kf[nt][kc],
                      Kb + (size_t)(kb * 64 + nt * 16 + nn) * DKK + kc * 32, h);

    float8 sfr[4];
#pragma unroll
    for (int nt = 0; nt < 4; ++nt) sfr[nt] = (float8)0.0f;
#pragma unroll
    for (int nt = 0; nt < 4; ++nt)
#pragma unroll
        for (int kc = 0; kc < 2; ++kc)
            sfr[nt] = wmma_f16(qf[kc], kf[nt][kc], sfr[nt]);

    // ---- V fragments: independent of softmax, issue now to hide latency ----
    HU vf[4][2];
#pragma unroll
    for (int nt = 0; nt < 4; ++nt)
#pragma unroll
        for (int kc = 0; kc < 2; ++kc)
            load_frag(vf[nt][kc],
                      Vb + (size_t)(nt * 16 + nn) * SS + kb * 64 + kc * 32, h);

    // ---- online softmax (scale already folded into Q) ----
#pragma unroll
    for (int r = 0; r < 8; ++r) {
        float mx = -3.0e38f;
#pragma unroll
        for (int nt = 0; nt < 4; ++nt) {
            float sv = sfr[nt][r];
            if (MASKED) {
                const int q = q0 + r + 8 * h;
                const int k = kb * 64 + nt * 16 + nn;
                if (k > q) sv = -1.0e9f;
                sfr[nt][r] = sv;
            }
            mx = fmaxf(mx, sv);
        }
        mx = fmaxf(mx, __shfl_xor(mx, 1));   // stays within 16-lane C half
        mx = fmaxf(mx, __shfl_xor(mx, 2));
        mx = fmaxf(mx, __shfl_xor(mx, 4));
        mx = fmaxf(mx, __shfl_xor(mx, 8));
        const float mnew  = fmaxf(m_run[r], mx);
        const float alpha = __expf(m_run[r] - mnew);
        l_run[r] *= alpha;
#pragma unroll
        for (int nt = 0; nt < 4; ++nt) o[nt][r] *= alpha;
        float rs = 0.0f;
#pragma unroll
        for (int nt = 0; nt < 4; ++nt) {
            const float p = __expf(sfr[nt][r] - mnew);
            sfr[nt][r] = p;
            rs += p;
        }
        rs += __shfl_xor(rs, 1);
        rs += __shfl_xor(rs, 2);
        rs += __shfl_xor(rs, 4);
        rs += __shfl_xor(rs, 8);
        l_run[r] += rs;
        m_run[r]  = mnew;
    }

    // ---- transpose P: C-layout f32 -> A-layout f16 via per-wave LDS ----
#pragma unroll
    for (int nt = 0; nt < 4; ++nt)
#pragma unroll
        for (int r = 0; r < 8; ++r)
            ldsP[(r + 8 * h) * PPAD + nt * 16 + nn] = (_Float16)sfr[nt][r];
    // same-wave LDS RAW ordered via DScnt (compiler-inserted wait)

    HU pf[2];
#pragma unroll
    for (int kc = 0; kc < 2; ++kc)
        load_frag(pf[kc], ldsP + nn * PPAD + kc * 32, h);

#pragma unroll
    for (int nt = 0; nt < 4; ++nt)
#pragma unroll
        for (int kc = 0; kc < 2; ++kc)
            o[nt] = wmma_f16(pf[kc], vf[nt][kc], o[nt]);
}

// ---------------------------------------------------------------------------
// Flash attention, causal. One wave owns 16 query rows end-to-end; no
// workgroup barriers (waves have different causal trip counts).
// Grid (S/128, H, B), block 256 (8 waves).
// ---------------------------------------------------------------------------
__global__ void __launch_bounds__(256)
attn_kernel(const _Float16* __restrict__ Qh, const _Float16* __restrict__ Kh,
            const _Float16* __restrict__ Vt, _Float16* __restrict__ ctx)
{
    __shared__ _Float16 lds_p[8][16][PPAD];

    const int lane = threadIdx.x & 31;
    const int wave = threadIdx.x >> 5;
    const int h    = lane >> 4;
    const int nn   = lane & 15;
    const int hh   = blockIdx.y;
    const int bb   = blockIdx.z;
    const int q0   = blockIdx.x * 128 + wave * 16;

    const _Float16* Qb = Qh + (size_t)(bb * HH + hh) * SS * DKK;
    const _Float16* Kb = Kh + (size_t)(bb * HH + hh) * SS * DKK;
    const _Float16* Vb = Vt + (size_t)(bb * HH + hh) * DKK * SS;
    _Float16* ldsP = &lds_p[wave][0][0];

    // Q fragments, with 1/sqrt(DK)=0.125 folded in (exact pow2 in f16).
    HU qf[2];
#pragma unroll
    for (int kc = 0; kc < 2; ++kc) {
        load_frag(qf[kc], Qb + (size_t)(q0 + nn) * DKK + kc * 32, h);
#pragma unroll
        for (int i = 0; i < 16; ++i)
            qf[kc].e[i] = qf[kc].e[i] * (_Float16)0.125f;
    }

    float8 o[4];
#pragma unroll
    for (int nt = 0; nt < 4; ++nt) o[nt] = (float8)0.0f;
    float m_run[8], l_run[8];
#pragma unroll
    for (int r = 0; r < 8; ++r) { m_run[r] = -3.0e38f; l_run[r] = 0.0f; }

    const int kbF = q0 / 64;             // full (unmasked) key blocks
#pragma unroll 1
    for (int kb = 0; kb < kbF; ++kb)
        attn_block<false>(kb, q0, h, nn, Kb, Vb, ldsP, qf, o, m_run, l_run);
    attn_block<true>(kbF, q0, h, nn, Kb, Vb, ldsP, qf, o, m_run, l_run);

    // ---- epilogue: normalize, store ctx[b, q, h*DK + dk] as f16 ----
#pragma unroll
    for (int nt = 0; nt < 4; ++nt) {
#pragma unroll
        for (int r = 0; r < 8; ++r) {
            const int q  = q0 + r + 8 * h;
            const int dk = nt * 16 + nn;
            const float val = o[nt][r] / l_run[r];
            ctx[((size_t)(bb * SS + q)) * DD + hh * DKK + dk] = (_Float16)val;
        }
    }
}

// ---------------------------------------------------------------------------
// Output projection: out[m,n] = sum_d ctx[m,d]*Wo[n,d] + bo[n], f32 out.
// Same double-buffered all-f16 structure as proj_kernel.
// ---------------------------------------------------------------------------
__global__ void __launch_bounds__(256)
outproj_kernel(const _Float16* __restrict__ Xh, const _Float16* __restrict__ Wh,
               const float* __restrict__ bias, float* __restrict__ out)
{
    const int lane = threadIdx.x & 31;
    const int wave = threadIdx.x >> 5;
    const int h    = lane >> 4;
    const int nn   = lane & 15;
    const int m0   = blockIdx.x * 128 + wave * 16;
    const int n0   = blockIdx.y * 64;

    const _Float16* xbase = Xh + (size_t)(m0 + nn) * DD;
    const _Float16* wbase[4];
#pragma unroll
    for (int nt = 0; nt < 4; ++nt)
        wbase[nt] = Wh + (size_t)(n0 + nt * 16 + nn) * DD;

    HU a[2], b[2][4];
    float8 acc[4];
#pragma unroll
    for (int nt = 0; nt < 4; ++nt) acc[nt] = (float8)0.0f;

    load_frag(a[0], xbase, h);
#pragma unroll
    for (int nt = 0; nt < 4; ++nt) load_frag(b[0][nt], wbase[nt], h);

#pragma unroll 1
    for (int kk = 0; kk < DD; kk += 64) {
        load_frag(a[1], xbase + kk + 32, h);
#pragma unroll
        for (int nt = 0; nt < 4; ++nt)
            load_frag(b[1][nt], wbase[nt] + kk + 32, h);
#pragma unroll
        for (int nt = 0; nt < 4; ++nt)
            acc[nt] = wmma_f16(a[0], b[0][nt], acc[nt]);

        if (kk + 64 < DD) {
            load_frag(a[0], xbase + kk + 64, h);
#pragma unroll
            for (int nt = 0; nt < 4; ++nt)
                load_frag(b[0][nt], wbase[nt] + kk + 64, h);
        }
#pragma unroll
        for (int nt = 0; nt < 4; ++nt)
            acc[nt] = wmma_f16(a[1], b[1][nt], acc[nt]);
    }

#pragma unroll
    for (int nt = 0; nt < 4; ++nt) {
        const int   n  = n0 + nt * 16 + nn;
        const float bv = bias[n];
#pragma unroll
        for (int r = 0; r < 8; ++r) {
            const int m = m0 + r + 8 * h;
            out[(size_t)m * DD + n] = acc[nt][r] + bv;
        }
    }
}

// ---------------------------------------------------------------------------
extern "C" void kernel_launch(void* const* d_in, const int* in_sizes, int n_in,
                              void* d_out, int out_size, void* d_ws, size_t ws_size,
                              hipStream_t stream)
{
    (void)in_sizes; (void)n_in; (void)out_size; (void)ws_size;

    const float* q  = (const float*)d_in[0];
    const float* k  = (const float*)d_in[1];
    const float* v  = (const float*)d_in[2];
    const float* wq = (const float*)d_in[3];
    const float* bq = (const float*)d_in[4];
    const float* wk = (const float*)d_in[5];
    const float* bk = (const float*)d_in[6];
    const float* wv = (const float*)d_in[7];
    const float* bv = (const float*)d_in[8];
    const float* wo = (const float*)d_in[9];
    const float* bo = (const float*)d_in[10];
    // d_in[11] = mask: causal, hardcoded in attn_kernel.

    char* ws = (char*)d_ws;
    const size_t PX = (size_t)BSROWS * DD * sizeof(_Float16);  // 12.58 MB
    const size_t PW = (size_t)DD * DD * sizeof(_Float16);      //  1.18 MB
    _Float16* xq  = (_Float16*)(ws);
    _Float16* xk  = (_Float16*)(ws + PX);
    _Float16* xv  = (_Float16*)(ws + 2 * PX);
    _Float16* wqh = (_Float16*)(ws + 3 * PX);
    _Float16* wkh = (_Float16*)(ws + 3 * PX + PW);
    _Float16* wvh = (_Float16*)(ws + 3 * PX + 2 * PW);
    _Float16* woh = (_Float16*)(ws + 3 * PX + 3 * PW);
    _Float16* qh  = (_Float16*)(ws + 3 * PX + 4 * PW);
    _Float16* kh  = (_Float16*)(ws + 4 * PX + 4 * PW);
    _Float16* vt  = (_Float16*)(ws + 5 * PX + 4 * PW);
    _Float16* ctx = (_Float16*)(ws + 6 * PX + 4 * PW);

    const int nx = BSROWS * DD;   // 6291456
    const int nw = DD * DD;       //  589824
    cvt_kernel<<<nx / 2048, 256, 0, stream>>>(q,  xq,  nx);
    cvt_kernel<<<nx / 2048, 256, 0, stream>>>(k,  xk,  nx);
    cvt_kernel<<<nx / 2048, 256, 0, stream>>>(v,  xv,  nx);
    cvt_kernel<<<nw / 2048, 256, 0, stream>>>(wq, wqh, nw);
    cvt_kernel<<<nw / 2048, 256, 0, stream>>>(wk, wkh, nw);
    cvt_kernel<<<nw / 2048, 256, 0, stream>>>(wv, wvh, nw);
    cvt_kernel<<<nw / 2048, 256, 0, stream>>>(wo, woh, nw);

    dim3 pg(BSROWS / 128, DD / 64);          // (64, 12)
    proj_kernel<<<pg, 256, 0, stream>>>(xq, wqh, bq, qh, 0);
    proj_kernel<<<pg, 256, 0, stream>>>(xk, wkh, bk, kh, 0);
    proj_kernel<<<pg, 256, 0, stream>>>(xv, wvh, bv, vt, 1);

    dim3 ag(SS / 128, HH, BB);               // (32, 12, 2)
    attn_kernel<<<ag, 256, 0, stream>>>(qh, kh, vt, ctx);

    outproj_kernel<<<pg, 256, 0, stream>>>(ctx, woh, bo, (float*)d_out);
}
